// DeformableConv2d_57543971832541
// MI455X (gfx1250) — matrix operand order repaired and checked by reading, not compiled, not run on hardware
//
#include <hip/hip_runtime.h>
#include <hip/hip_bf16.h>

typedef __attribute__((ext_vector_type(16))) _Float16 v16h;
typedef __attribute__((ext_vector_type(8)))  _Float16 v8h;
typedef __attribute__((ext_vector_type(4)))  _Float16 v4h;
typedef __attribute__((ext_vector_type(2)))  _Float16 v2h;
typedef __attribute__((ext_vector_type(8)))  float    v8f;
typedef __attribute__((ext_vector_type(4)))  int      v4i;

#define B_    8
#define C_    112
#define H_    64
#define W_    64
#define O_    112
#define G_    14
#define K2_   9
#define GK2_  126
#define PGCH  378      // 3*G*K2
#define HW_   4096
#define KRED  1008     // C_*K2_
#define KPAD  1024     // padded reduction length (zero-filled tail)
#define MPAD  384      // padded pg-conv M (24 tiles of 16)
#define BSTRIDE 40     // halves per B-tile row in LDS (80B, 16B-aligned, padded)

#if __has_builtin(__builtin_amdgcn_global_load_async_to_lds_b128) && \
    __has_builtin(__builtin_amdgcn_s_wait_asynccnt)
#define USE_ASYNC 1
typedef __attribute__((address_space(1))) v4i* gv4i_p;   // global b128 pointer
typedef __attribute__((address_space(3))) v4i* lv4i_p;   // LDS b128 pointer
#else
#define USE_ASYNC 0
#endif

struct __align__(16) Smp {            // one bilinear sample descriptor (16B)
    v4h      w;                       // mask-premultiplied, validity-zeroed weights
    unsigned crd;                     // packed clamped corner coords
    unsigned pad;
};

// ---------------------------------------------------------------------------
// Kernel 0: convert operands to f16 with K padded to 1024 (zero tail)
// ---------------------------------------------------------------------------
__global__ __launch_bounds__(256)
void cvt_kernel(const float* __restrict__ w, const float* __restrict__ pgw,
                const float* __restrict__ x,
                _Float16* __restrict__ w16, _Float16* __restrict__ pgw16,
                _Float16* __restrict__ x16) {
    const int i = blockIdx.x * 256 + threadIdx.x;
    const int stride = gridDim.x * 256;
    const int nx4 = (B_ * C_ * HW_) / 4;          // 917504
    const float4* xs4 = (const float4*)x;
    const float4* ws4 = (const float4*)w;
    const float4* ps4 = (const float4*)pgw;
    v4h* xd = (v4h*)x16;
    v4h* wd = (v4h*)w16;      // [112][256] v4h rows
    v4h* pd = (v4h*)pgw16;    // [384][256] v4h rows
    const v4h z4 = {};

    for (int t = i; t < nx4; t += stride) {
        float4 f = xs4[t];
        v4h h; h[0]=(_Float16)f.x; h[1]=(_Float16)f.y; h[2]=(_Float16)f.z; h[3]=(_Float16)f.w;
        xd[t] = h;
    }
    for (int t = i; t < O_ * 252; t += stride) {
        const int r = t / 252, c4 = t - r * 252;
        float4 f = ws4[t];
        v4h h; h[0]=(_Float16)f.x; h[1]=(_Float16)f.y; h[2]=(_Float16)f.z; h[3]=(_Float16)f.w;
        wd[r * 256 + c4] = h;
    }
    for (int t = i; t < O_ * 4; t += stride)
        wd[(t >> 2) * 256 + 252 + (t & 3)] = z4;
    for (int t = i; t < PGCH * 252; t += stride) {
        const int r = t / 252, c4 = t - r * 252;
        float4 f = ps4[t];
        v4h h; h[0]=(_Float16)f.x; h[1]=(_Float16)f.y; h[2]=(_Float16)f.z; h[3]=(_Float16)f.w;
        pd[r * 256 + c4] = h;
    }
    for (int t = i; t < PGCH * 4; t += stride)
        pd[(t >> 2) * 256 + 252 + (t & 3)] = z4;
    for (int t = i; t < 6 * 256; t += stride)
        pd[PGCH * 256 + t] = z4;
}

// ---------------------------------------------------------------------------
// Kernel 1: pg = conv3x3(x, pg_weight) + pg_bias   (implicit GEMM, WMMA f16)
//   M = 378 (pad 384 = 24 tiles), N = 32768 pixels, K = 1008 (pad 1024)
//   Block: 256 thr / 8 waves; M_block = 384 (3 tiles/wave), N_block = 16.
//   Each staged B element feeds 24 WMMAs -> 6x less im2col work than M_blk=64.
// ---------------------------------------------------------------------------
__global__ __launch_bounds__(256)
void pg_conv_kernel(const _Float16* __restrict__ pgw16,
                    const _Float16* __restrict__ x16,
                    const float* __restrict__ pg_bias,
                    float* __restrict__ pg) {
    __shared__ _Float16 Bfrag[16][BSTRIDE];   // [n][k] fragment-order, padded
    __shared__ unsigned s_ckr[KPAD];          // (c<<8)|(ky<<4)|kx ; bit31 = pad-k

    const int tid    = threadIdx.x;
    const int lane   = tid & 31;
    const int wave   = tid >> 5;        // 0..7
    const int lane15 = lane & 15;
    const int halfId = lane >> 4;

    const int n_block = blockIdx.x * 16;

    for (int k = tid; k < KPAD; k += 256) {
        unsigned v = 0x80000000u;       // sentinel: pad-k
        if (k < KRED) {
            const int c  = k / 9;
            const int t9 = k - c * 9;
            const int ky = t9 / 3;
            const int kx = t9 - ky * 3;
            v = ((unsigned)c << 8) | ((unsigned)ky << 4) | (unsigned)kx;
        }
        s_ckr[k] = v;
    }

    // staging mapping: thread -> (2 consecutive k, one n)
    const int nl  = tid & 15;
    const int klb = (tid >> 4) * 2;     // 0,2,..,30
    const int n   = n_block + nl;
    const int bb  = n >> 12;
    const int hw  = n & 4095;
    const int hh  = hw >> 6;
    const int ww  = hw & 63;
    const _Float16* xplane = x16 + (size_t)bb * C_ * HW_;

    // three M tiles per wave: tiles wave, wave+8, wave+16
    const _Float16* arow0 = pgw16 + (size_t)(((wave     ) << 4) + lane15) * KPAD;
    const _Float16* arow1 = pgw16 + (size_t)(((wave +  8) << 4) + lane15) * KPAD;
    const _Float16* arow2 = pgw16 + (size_t)(((wave + 16) << 4) + lane15) * KPAD;

    v8f acc0 = {}, acc1 = {}, acc2 = {};
    __syncthreads();   // LUT ready

    for (int k0 = 0; k0 < KPAD; k0 += 32) {
        // ---- stage im2col B chunk (32k x 16n): branchless, 2 values/thread ----
        v2h pk;
        #pragma unroll
        for (int i = 0; i < 2; ++i) {
            const unsigned d = s_ckr[k0 + klb + i];
            const int c  = (int)((d >> 8) & 0x7Fu);
            const int ky = (int)((d >> 4) & 15u);
            const int kx = (int)(d & 15u);
            const int y  = hh + ky - 1;
            const int xx = ww + kx - 1;
            const int yc = min(max(y, 0), 63);
            const int xc = min(max(xx, 0), 63);
            const _Float16 ld = xplane[(c << 12) + (yc << 6) + xc];
            const bool ok = ((unsigned)y < 64u) && ((unsigned)xx < 64u) && (d < 0x80000000u);
            pk[i] = ok ? ld : (_Float16)0.f;
        }
        *(v2h*)&Bfrag[nl][klb] = pk;     // single ds_store_b32
        __syncthreads();

        // ---- B fragment (shared by all 3 M tiles): two ds_load_b128 ----
        const v8h* bp = (const v8h*)&Bfrag[lane15][8 * halfId];
        v8h b0 = bp[0];
        v8h b1 = bp[2];
        v16h bf = __builtin_shufflevector(b0, b1, 0,1,2,3,4,5,6,7,8,9,10,11,12,13,14,15);

        // ---- 3 A fragments + 3 WMMAs ----
        {
            const v8h* ap = (const v8h*)(arow0 + k0 + 8 * halfId);
            v16h a = __builtin_shufflevector(ap[0], ap[2], 0,1,2,3,4,5,6,7,8,9,10,11,12,13,14,15);
            acc0 = __builtin_amdgcn_wmma_f32_16x16x32_f16(false, a, false, bf, (short)0, acc0, false, false);
        }
        {
            const v8h* ap = (const v8h*)(arow1 + k0 + 8 * halfId);
            v16h a = __builtin_shufflevector(ap[0], ap[2], 0,1,2,3,4,5,6,7,8,9,10,11,12,13,14,15);
            acc1 = __builtin_amdgcn_wmma_f32_16x16x32_f16(false, a, false, bf, (short)0, acc1, false, false);
        }
        {
            const v8h* ap = (const v8h*)(arow2 + k0 + 8 * halfId);
            v16h a = __builtin_shufflevector(ap[0], ap[2], 0,1,2,3,4,5,6,7,8,9,10,11,12,13,14,15);
            acc2 = __builtin_amdgcn_wmma_f32_16x16x32_f16(false, a, false, bf, (short)0, acc2, false, false);
        }
        if (k0 + 64 <= KPAD) {
            __builtin_prefetch(arow0 + k0 + 32, 0, 3);   // global_prefetch_b8
            __builtin_prefetch(arow1 + k0 + 32, 0, 3);
            __builtin_prefetch(arow2 + k0 + 32, 0, 3);
        }
        __syncthreads();
    }

    // ---- epilogue: bias + store pg[b][ch][hw] for 3 tiles ----
    const int nout = n_block + lane15;
    const int ob   = nout >> 12;
    const int ohw  = nout & 4095;
    #pragma unroll
    for (int r = 0; r < 8; ++r) {
        const int rr = r + 8 * halfId;
        const int m0 = ((wave     ) << 4) + rr;
        const int m1 = ((wave +  8) << 4) + rr;
        const int m2 = ((wave + 16) << 4) + rr;
        pg[((size_t)ob * PGCH + m0) * HW_ + ohw] = acc0[r] + pg_bias[m0];
        pg[((size_t)ob * PGCH + m1) * HW_ + ohw] = acc1[r] + pg_bias[m1];
        if (m2 < PGCH)
            pg[((size_t)ob * PGCH + m2) * HW_ + ohw] = acc2[r] + pg_bias[m2];
    }
}

// ---------------------------------------------------------------------------
// Kernel 2: fused bilinear sampling + deformable einsum  (WMMA f16)
//   Block: 256 thr / 8 waves; waves 0..6 own the 7 M tiles, all stage.
// ---------------------------------------------------------------------------
__global__ __launch_bounds__(256)
void dcn_kernel(const float* __restrict__ x, const float* __restrict__ pg,
                const _Float16* __restrict__ w16, const float* __restrict__ bias,
                float* __restrict__ out) {
    __shared__ float    s_pg[PGCH][16];      // async-staged pg rows (24.2 KB)
    __shared__ Smp      s_smp[GK2_ * 16];    // sample descriptors (31.5 KB)
    __shared__ unsigned s_ckr[KPAD];         // (c<<16)|p-base ; bit31 = pad-k
    __shared__ _Float16 Bfrag[16][BSTRIDE];  // [n][k] fragment-order

    const int tid    = threadIdx.x;
    const int lane   = tid & 31;
    const int wave   = tid >> 5;        // waves 0..6 -> M tiles
    const int lane15 = lane & 15;
    const int halfId = lane >> 4;

    const int n0  = blockIdx.x * 16;
    const int bb  = n0 >> 12;
    const int hw0 = n0 & 4095;
    const int hh  = hw0 >> 6;
    const int ww0 = hw0 & 63;
    const float* xbatch = x + (size_t)bb * C_ * HW_;

    // ---- async-stage the 378 pg rows (64B each) this block needs ----
    const float* pgbase = pg + (size_t)bb * PGCH * HW_ + hw0;
    for (int t = tid; t < PGCH * 4; t += 256) {
        const int row  = t >> 2;
        const int part = (t & 3) << 2;                 // float offset in row
        const float* src = pgbase + (size_t)row * HW_ + part;
#if USE_ASYNC
        __builtin_amdgcn_global_load_async_to_lds_b128(
            (gv4i_p)src, (lv4i_p)&s_pg[row][part], 0, 0);
#else
        *(float4*)&s_pg[row][part] = *(const float4*)src;
#endif
    }

    // ---- decode LUT ----
    for (int k = tid; k < KPAD; k += 256) {
        unsigned v = 0x80000000u;       // sentinel: pad-k (c=0, p-base=0: safe)
        if (k < KRED) {
            const int c = k / 9;
            const int r = k - c * 9;
            const int g = c >> 3;                      // C/G = 8
            v = ((unsigned)c << 16) | (unsigned)((g * 9 + r) * 16);
        }
        s_ckr[k] = v;
    }

#if USE_ASYNC
    __builtin_amdgcn_s_wait_asynccnt(0);
#endif
    __syncthreads();

    // ---- precompute sample descriptors for 126 (g,k2) x 16 pixels ----
    for (int p = tid; p < GK2_ * 16; p += 256) {
        const int gk = p >> 4;
        const int nl = p & 15;
        const int g  = gk / 9;
        const int r  = gk - g * 9;
        const int ky = r / 3;
        const int kx = r - ky * 3;
        const float dy = s_pg[gk][nl];
        const float dx = s_pg[GK2_ + gk][nl];
        const float mm = s_pg[2 * GK2_ + gk][nl];
        const float ys = (float)(hh + ky - 1) + dy;
        const float xs = (float)(ww0 + nl + kx - 1) + dx;
        const float fy0 = floorf(ys);
        const float fx0 = floorf(xs);
        const int iy0 = (int)fy0;
        const int ix0 = (int)fx0;
        const float fy = ys - fy0, fx = xs - fx0;
        const float mk = 1.0f / (1.0f + __expf(-mm));
        const float wy0 = 1.f - fy, wx0 = 1.f - fx;
        const bool y0ok = (unsigned)iy0       < 64u;
        const bool y1ok = (unsigned)(iy0 + 1) < 64u;
        const bool x0ok = (unsigned)ix0       < 64u;
        const bool x1ok = (unsigned)(ix0 + 1) < 64u;
        Smp sm;
        sm.w[0] = (_Float16)((y0ok && x0ok) ? wy0 * wx0 * mk : 0.f);
        sm.w[1] = (_Float16)((y0ok && x1ok) ? wy0 * fx  * mk : 0.f);
        sm.w[2] = (_Float16)((y1ok && x0ok) ? fy  * wx0 * mk : 0.f);
        sm.w[3] = (_Float16)((y1ok && x1ok) ? fy  * fx  * mk : 0.f);
        const int iy0c = min(max(iy0, 0), 63);
        const int iy1c = min(max(iy0 + 1, 0), 63);
        const int ix0c = min(max(ix0, 0), 63);
        const int ix1c = min(max(ix0 + 1, 0), 63);
        sm.crd = (unsigned)((iy0c << 18) | (iy1c << 12) | (ix0c << 6) | ix1c);
        sm.pad = 0u;
        s_smp[p] = sm;                                 // single ds_store_b128
    }
    __syncthreads();

    v8f acc = {};
    const int m = wave * 16 + lane15;                  // valid for waves 0..6
    const _Float16* arow = w16 + (size_t)min(m, O_ - 1) * KPAD;

    for (int k0 = 0; k0 < KPAD; k0 += 32) {
        // ---- sample B chunk (32k x 16n): exactly 2 values/thread, branchless ----
        #pragma unroll
        for (int it = 0; it < 2; ++it) {
            const int v  = tid + it * 256;
            const int kl = v >> 4;
            const int nl = v & 15;
            const unsigned d = s_ckr[k0 + kl];
            const int c = (int)((d >> 16) & 0x7FFFu);
            const int p = (int)(d & 0xFFFFu) + nl;
            const Smp sm = s_smp[p];                   // one ds_load_b128
            const int row0 = ((sm.crd >> 18) & 63) << 6;
            const int row1 = ((sm.crd >> 12) & 63) << 6;
            const int col0 = (sm.crd >> 6) & 63;
            const int col1 = sm.crd & 63;
            const float* xb = xbatch + (c << 12);
            const float v00 = xb[row0 + col0];
            const float v01 = xb[row0 + col1];
            const float v10 = xb[row1 + col0];
            const float v11 = xb[row1 + col1];
            float val = v00 * (float)sm.w[0] + v01 * (float)sm.w[1]
                      + v10 * (float)sm.w[2] + v11 * (float)sm.w[3];
            val = (d >> 31) ? 0.f : val;
            Bfrag[nl][kl] = (_Float16)val;
        }
        __syncthreads();

        if (wave < 7) {
            const v8h* ap = (const v8h*)(arow + k0 + 8 * halfId);
            v8h lo = ap[0];
            v8h hi = ap[2];
            v16h a = __builtin_shufflevector(lo, hi, 0,1,2,3,4,5,6,7,8,9,10,11,12,13,14,15);
            if (k0 + 64 <= KPAD)
                __builtin_prefetch(arow + k0 + 32, 0, 3);   // global_prefetch_b8

            const v8h* bp = (const v8h*)&Bfrag[lane15][8 * halfId];
            v8h b0 = bp[0];
            v8h b1 = bp[2];
            v16h bf = __builtin_shufflevector(b0, b1, 0,1,2,3,4,5,6,7,8,9,10,11,12,13,14,15);

            acc = __builtin_amdgcn_wmma_f32_16x16x32_f16(
                      false, a, false, bf, (short)0, acc, false, false);
        }
        __syncthreads();
    }

    if (wave < 7) {
        #pragma unroll
        for (int r = 0; r < 8; ++r) {
            const int o = wave * 16 + r + 8 * halfId;
            out[((size_t)bb * O_ + o) * HW_ + hw0 + lane15] = acc[r] + bias[o];
        }
    }
}

// ---------------------------------------------------------------------------
extern "C" void kernel_launch(void* const* d_in, const int* in_sizes, int n_in,
                              void* d_out, int out_size, void* d_ws, size_t ws_size,
                              hipStream_t stream) {
    (void)in_sizes; (void)n_in; (void)out_size; (void)ws_size;

    const float* x   = (const float*)d_in[0];
    const float* pgw = (const float*)d_in[1];
    const float* pgb = (const float*)d_in[2];
    const float* w   = (const float*)d_in[3];
    const float* bs  = (const float*)d_in[4];

    char* ws = (char*)d_ws;
    size_t off = 0;
    float* pg = (float*)(ws + off);
    off += (size_t)B_ * PGCH * HW_ * sizeof(float);          // 49.5 MB
    off = (off + 255) & ~(size_t)255;
    _Float16* w16 = (_Float16*)(ws + off);
    off += (size_t)O_ * KPAD * sizeof(_Float16);             // 0.23 MB
    off = (off + 255) & ~(size_t)255;
    _Float16* pgw16 = (_Float16*)(ws + off);
    off += (size_t)MPAD * KPAD * sizeof(_Float16);           // 0.79 MB
    off = (off + 255) & ~(size_t)255;
    _Float16* x16 = (_Float16*)(ws + off);
    off += (size_t)B_ * C_ * HW_ * sizeof(_Float16);         // 7.3 MB

    cvt_kernel<<<3584, 256, 0, stream>>>(w, pgw, x, w16, pgw16, x16);
    pg_conv_kernel<<<2048, 256, 0, stream>>>(pgw16, x16, pgb, pg);
    dcn_kernel<<<2048, 256, 0, stream>>>(x, pg, w16, bs, (float*)d_out);
}
